// Attention_40561671143650
// MI455X (gfx1250) — compile-verified, matrix operand order checked by reading, not run on hardware
//
#include <hip/hip_runtime.h>

// ---- problem constants (from reference) ----
#define SLEN 2048
#define HID  2048
#define NHQ  32
#define NKV  8
#define HD   64
#define QSZ  2048          // NHQ*HD
#define KVSZ 512           // NKV*HD
#define QKVO 3072          // QSZ + 2*KVSZ

typedef __attribute__((ext_vector_type(16))) __bf16 v16bf;
typedef __attribute__((ext_vector_type(8)))  float  v8f;

union ABVec {
    v16bf  v;
    uint4  q[2];
    __bf16 e[16];
};

__device__ __forceinline__ float wave16_max(float x) {
#pragma unroll
    for (int s = 1; s < 16; s <<= 1) x = fmaxf(x, __shfl_xor(x, s, 32));
    return x;
}
__device__ __forceinline__ float wave16_sum(float x) {
#pragma unroll
    for (int s = 1; s < 16; s <<= 1) x += __shfl_xor(x, s, 32);
    return x;
}

// ---------------- f32 -> bf16 conversion ----------------
__global__ void cvt_bf16_kernel(const float* __restrict__ in,
                                __bf16* __restrict__ out, int n) {
    int i = blockIdx.x * blockDim.x + threadIdx.x;
    if (i < n) out[i] = (__bf16)in[i];
}

// ---------------- WMMA GEMM: C[M,N] = A[M,K] * W[N,K]^T ----------------
// One wave computes a 16(M) x 64(N) tile: 4 accumulators, A reused 4x.
// bf16 A/B layout per lane (hi = lane>>4): element e -> K index
//   k(e) = (e>=8 ? 16 : 0) + hi*8 + (e&7)  => two contiguous 16B chunks.
__global__ __launch_bounds__(256) void gemm_bf16_kernel(
    const __bf16* __restrict__ A, const __bf16* __restrict__ W,
    float* __restrict__ C, int M, int N, int K) {
    int wave   = (int)((blockIdx.x * 256u + threadIdx.x) >> 5);
    int lane   = threadIdx.x & 31;
    int tilesN = N >> 6;
    int tm = wave / tilesN;
    int tn = wave % tilesN;
    if (tm >= (M >> 4)) return;
    int m  = lane & 15;
    int hi = lane >> 4;

    const __bf16* arow  = A + (size_t)(tm * 16 + m) * K;
    const __bf16* wrow0 = W + (size_t)(tn * 64 + m) * K;

    v8f acc[4] = {};
    for (int k0 = 0; k0 < K; k0 += 32) {
        ABVec a;
        a.q[0] = *reinterpret_cast<const uint4*>(arow + k0 + hi * 8);
        a.q[1] = *reinterpret_cast<const uint4*>(arow + k0 + 16 + hi * 8);
#pragma unroll
        for (int t = 0; t < 4; ++t) {
            const __bf16* wr = wrow0 + (size_t)t * 16 * K;
            ABVec b;
            b.q[0] = *reinterpret_cast<const uint4*>(wr + k0 + hi * 8);
            b.q[1] = *reinterpret_cast<const uint4*>(wr + k0 + 16 + hi * 8);
            acc[t] = __builtin_amdgcn_wmma_f32_16x16x32_bf16(
                false, a.v, false, b.v, (short)0, acc[t], false, false);
        }
    }
#pragma unroll
    for (int t = 0; t < 4; ++t)
#pragma unroll
        for (int r = 0; r < 8; ++r)
            C[(size_t)(tm * 16 + r + 8 * hi) * N + tn * 64 + t * 16 + m] = acc[t][r];
}

// ---------------- RoPE (neox) + split qkv -> bf16 q/k/v ----------------
__global__ void rope_split_kernel(const float* __restrict__ qkv,
                                  __bf16* __restrict__ qo,
                                  __bf16* __restrict__ ko,
                                  __bf16* __restrict__ vo) {
    const int QP = SLEN * NHQ * (HD / 2);
    const int KP = SLEN * NKV * (HD / 2);
    const int VN = SLEN * KVSZ;
    int idx = blockIdx.x * blockDim.x + threadIdx.x;
    if (idx < QP) {
        int i = idx & 31;
        int t = idx >> 5;
        int h = t % NHQ;
        int s = t / NHQ;
        const float* src = qkv + (size_t)s * QKVO + h * HD;
        float x1 = src[i], x2 = src[i + 32];
        float freq = __powf(10000.0f, -(float)i / 32.0f);
        float sn, cs;
        __sincosf((float)s * freq, &sn, &cs);
        __bf16* dst = qo + (size_t)s * QSZ + h * HD;
        dst[i]      = (__bf16)(x1 * cs - x2 * sn);
        dst[i + 32] = (__bf16)(x2 * cs + x1 * sn);
    } else if (idx < QP + KP) {
        int j = idx - QP;
        int i = j & 31;
        int t = j >> 5;
        int h = t % NKV;
        int s = t / NKV;
        const float* src = qkv + (size_t)s * QKVO + QSZ + h * HD;
        float x1 = src[i], x2 = src[i + 32];
        float freq = __powf(10000.0f, -(float)i / 32.0f);
        float sn, cs;
        __sincosf((float)s * freq, &sn, &cs);
        __bf16* dst = ko + (size_t)s * KVSZ + h * HD;
        dst[i]      = (__bf16)(x1 * cs - x2 * sn);
        dst[i + 32] = (__bf16)(x2 * cs + x1 * sn);
    } else if (idx < QP + KP + VN) {
        int j = idx - QP - KP;
        int s = j / KVSZ, c = j % KVSZ;
        vo[(size_t)s * KVSZ + c] = (__bf16)qkv[(size_t)s * QKVO + QSZ + KVSZ + c];
    }
}

// ---------------- Flash attention: one wave per (head, 16-row q tile) ----------------
__global__ __launch_bounds__(32) void attn_kernel(
    const __bf16* __restrict__ q, const __bf16* __restrict__ k,
    const __bf16* __restrict__ v, __bf16* __restrict__ ctx) {
    __shared__ __align__(16) __bf16 vlds[32 * HD];  // V chunk, [local_key][d]
    __shared__ __align__(16) __bf16 plds[16 * 32];  // P tile,  [q][local_key]

    int h   = blockIdx.x / (SLEN / 16);
    int tq  = blockIdx.x % (SLEN / 16);
    int qb0 = tq * 16;
    int lane = threadIdx.x, m = lane & 15, hi = lane >> 4;
    int kvh = h >> 2;  // GQA: 4 query heads per kv head

    // Q tile, A-layout for both d-halves (K=32 each)
    const __bf16* qrow = q + (size_t)(qb0 + m) * QSZ + h * HD;
    ABVec aq[2];
    aq[0].q[0] = *reinterpret_cast<const uint4*>(qrow + hi * 8);
    aq[0].q[1] = *reinterpret_cast<const uint4*>(qrow + 16 + hi * 8);
    aq[1].q[0] = *reinterpret_cast<const uint4*>(qrow + 32 + hi * 8);
    aq[1].q[1] = *reinterpret_cast<const uint4*>(qrow + 48 + hi * 8);

    v8f o[4] = {};
    float mx[8], lsum[8];
#pragma unroll
    for (int r = 0; r < 8; ++r) { mx[r] = -1e30f; lsum[r] = 0.0f; }

    int kend = qb0 + 16;  // keys visible to the last row of the tile
    for (int kb0 = 0; kb0 < kend; kb0 += 32) {
        // ---- scores: two 16x16 key tiles via 2 WMMAs each ----
        int kr0 = kb0 + m;      if (kr0 > SLEN - 1) kr0 = SLEN - 1;
        int kr1 = kb0 + 16 + m; if (kr1 > SLEN - 1) kr1 = SLEN - 1;
        const __bf16* kp0 = k + (size_t)kr0 * KVSZ + kvh * HD;
        const __bf16* kp1 = k + (size_t)kr1 * KVSZ + kvh * HD;
        v8f s0 = {}, s1 = {};
#pragma unroll
        for (int dh = 0; dh < 2; ++dh) {
            ABVec b0, b1;
            b0.q[0] = *reinterpret_cast<const uint4*>(kp0 + dh * 32 + hi * 8);
            b0.q[1] = *reinterpret_cast<const uint4*>(kp0 + dh * 32 + 16 + hi * 8);
            b1.q[0] = *reinterpret_cast<const uint4*>(kp1 + dh * 32 + hi * 8);
            b1.q[1] = *reinterpret_cast<const uint4*>(kp1 + dh * 32 + 16 + hi * 8);
            s0 = __builtin_amdgcn_wmma_f32_16x16x32_bf16(
                false, aq[dh].v, false, b0.v, (short)0, s0, false, false);
            s1 = __builtin_amdgcn_wmma_f32_16x16x32_bf16(
                false, aq[dh].v, false, b1.v, (short)0, s1, false, false);
        }

        // ---- stage V chunk rows into LDS (lane <-> local key) ----
        int vr = kb0 + lane; if (vr > SLEN - 1) vr = SLEN - 1;
        const __bf16* vp = v + (size_t)vr * KVSZ + kvh * HD;
#pragma unroll
        for (int j = 0; j < 8; ++j)
            *reinterpret_cast<uint4*>(&vlds[lane * HD + j * 8]) =
                *reinterpret_cast<const uint4*>(vp + j * 8);

        // ---- online softmax (rows r+8*hi live across lanes with same hi) ----
#pragma unroll
        for (int r = 0; r < 8; ++r) {
            int qi = qb0 + r + 8 * hi;
            float x0 = s0[r] * 0.125f;  // 1/sqrt(64)
            float x1 = s1[r] * 0.125f;
            if (kb0 + m > qi)      x0 = -1e30f;  // causal mask
            if (kb0 + 16 + m > qi) x1 = -1e30f;
            float rm   = wave16_max(fmaxf(x0, x1));
            float mnew = fmaxf(mx[r], rm);
            float alpha = __expf(mx[r] - mnew);
            mx[r] = mnew;
            float p0 = __expf(x0 - mnew);
            float p1 = __expf(x1 - mnew);
            lsum[r] = lsum[r] * alpha + wave16_sum(p0 + p1);
#pragma unroll
            for (int t = 0; t < 4; ++t) o[t][r] *= alpha;
            plds[(r + 8 * hi) * 32 + m]      = (__bf16)p0;
            plds[(r + 8 * hi) * 32 + 16 + m] = (__bf16)p1;
        }
        __syncthreads();

        // ---- P in A-layout from LDS (row m, two contiguous 16B chunks) ----
        ABVec ap;
        ap.q[0] = *reinterpret_cast<const uint4*>(&plds[m * 32 + hi * 8]);
        ap.q[1] = *reinterpret_cast<const uint4*>(&plds[m * 32 + 16 + hi * 8]);

        // ---- O += P(16x32) * V(32x64): 4 N-tiles of 16 dims ----
#pragma unroll
        for (int t = 0; t < 4; ++t) {
            ABVec bv;
#pragma unroll
            for (int e = 0; e < 16; ++e) {
                int kk = ((e >> 3) * 16) + hi * 8 + (e & 7);
                bv.e[e] = vlds[kk * HD + t * 16 + m];
            }
            o[t] = __builtin_amdgcn_wmma_f32_16x16x32_bf16(
                false, ap.v, false, bv.v, (short)0, o[t], false, false);
        }
        __syncthreads();
    }

    // ---- normalize and store ctx as bf16 [s, h, d] ----
#pragma unroll
    for (int t = 0; t < 4; ++t)
#pragma unroll
        for (int r = 0; r < 8; ++r) {
            float val = o[t][r] / lsum[r];
            ctx[(size_t)(qb0 + r + 8 * hi) * QSZ + h * HD + t * 16 + m] = (__bf16)val;
        }
}

// ---------------- host-side orchestration ----------------
extern "C" void kernel_launch(void* const* d_in, const int* in_sizes, int n_in,
                              void* d_out, int out_size, void* d_ws, size_t ws_size,
                              hipStream_t stream) {
    (void)in_sizes; (void)n_in; (void)out_size; (void)ws_size;
    // inputs: 0=position_ids (arange, unused), 1=hidden_states f32 [S,H],
    //         2=w_qkv f32 [3072,2048], 3=w_o f32 [2048,2048]
    const float* hs   = (const float*)d_in[1];
    const float* wqkv = (const float*)d_in[2];
    const float* wo   = (const float*)d_in[3];
    float* out = (float*)d_out;

    char* ws = (char*)d_ws;
    // region 0 (8 MiB): xb, later reused as q_bf16 (same size)
    __bf16* xb = (__bf16*)ws;
    __bf16* qb = (__bf16*)ws;
    // region 1 (12 MiB): wqkv_bf16, later reused as k_bf16 + v_bf16 + wo_bf16
    char* r1 = ws + (size_t)SLEN * HID * 2;
    __bf16* wqkvb = (__bf16*)r1;
    __bf16* kb    = (__bf16*)r1;
    __bf16* vb    = (__bf16*)(r1 + (size_t)SLEN * KVSZ * 2);
    __bf16* wob   = (__bf16*)(r1 + (size_t)2 * SLEN * KVSZ * 2);
    // region 2 (24 MiB): qkv f32, later reused as ctx_bf16
    char* r2 = r1 + (size_t)QKVO * HID * 2;
    float*  qkvf = (float*)r2;
    __bf16* ctxb = (__bf16*)r2;

    // 1) convert activations + qkv weight to bf16
    {
        int n = SLEN * HID;
        cvt_bf16_kernel<<<(n + 255) / 256, 256, 0, stream>>>(hs, xb, n);
        n = QKVO * HID;
        cvt_bf16_kernel<<<(n + 255) / 256, 256, 0, stream>>>(wqkv, wqkvb, n);
    }
    // 2) QKV projection: [S,3072] = x[S,2048] * w_qkv^T
    {
        int tiles = (SLEN / 16) * (QKVO / 64);
        gemm_bf16_kernel<<<(tiles + 7) / 8, 256, 0, stream>>>(
            xb, wqkvb, qkvf, SLEN, QKVO, HID);
    }
    // 3) RoPE + split into bf16 q/k/v (overwrites xb/wqkvb regions — now dead)
    {
        int n = SLEN * (NHQ * 32 + NKV * 32 + KVSZ);
        rope_split_kernel<<<(n + 255) / 256, 256, 0, stream>>>(qkvf, qb, kb, vb);
    }
    // 4) convert w_o to bf16 (into tail of region 1)
    {
        int n = HID * QSZ;
        cvt_bf16_kernel<<<(n + 255) / 256, 256, 0, stream>>>(wo, wob, n);
    }
    // 5) causal GQA flash attention -> ctx bf16 (overwrites qkvf — now dead)
    attn_kernel<<<NHQ * (SLEN / 16), 32, 0, stream>>>(qb, kb, vb, ctxb);
    // 6) output projection: out[S,2048] = ctx[S,2048] * w_o^T (f32 out)
    {
        int tiles = (SLEN / 16) * (HID / 64);
        gemm_bf16_kernel<<<(tiles + 7) / 8, 256, 0, stream>>>(
            ctxb, wob, out, SLEN, HID, QSZ);
    }
}
